// AMIPRouterTrain_14559939133631
// MI455X (gfx1250) — matrix operand order repaired and checked by reading, compiled, not run on hardware
//
#include <hip/hip_runtime.h>
#include <hip/hip_bf16.h>

#define D_MODEL 4096
#define KEXP    8
#define D_HID   2048
#define D_PROJ  512
#define NTOK    4096

// ---- vector types matching gfx1250 WMMA operand shapes ----
typedef __bf16 v16bf __attribute__((ext_vector_type(16)));
typedef __bf16 bf8_t __attribute__((ext_vector_type(8)));
typedef __bf16 bf4_t __attribute__((ext_vector_type(4)));
typedef float  v8f   __attribute__((ext_vector_type(8)));
typedef float  f4_t  __attribute__((ext_vector_type(4)));

__device__ __forceinline__ __bf16 f2bf(float f) {
  // round-to-nearest-even fp32 -> bf16
  unsigned u = __builtin_bit_cast(unsigned, f);
  unsigned r = u + 0x7fffu + ((u >> 16) & 1u);
  return __builtin_bit_cast(__bf16, (unsigned short)(r >> 16));
}

__device__ __forceinline__ float gelu_exact(float x) {
  return 0.5f * x * (1.0f + erff(x * 0.70710678118654752f));
}

// =====================================================================
// Tiled bf16 WMMA GEMM: C[M,Ndim] (op)= A[M,Kd] * B[Kd,Ndim]
//   Block tile 128x128, 256 threads = 8 waves (2x4), wave tile 64x32,
//   K-step 32 -> v_wmma_f32_16x16x32_bf16, single-barrier double buffer.
// A is ALWAYS bf16 row-major (packed cond / hid).
// BTB=true : B is bf16 PRE-TRANSPOSED [Ndim][Kd] -> pure b64 copy staging.
// BTB=false: B is fp32 row-major [Kd][Ndim] -> convert + scatter (fallback).
// LDS pitch 40 halves (80B) -> bank spread + 16B-aligned frag loads.
//
// MODE 0 (PROJ): epi: acc + bias              -> fp32 C
// MODE 1 (HID):  epi: gelu(acc+b1)*w[n,kexp]  -> bf16 C
// MODE 2 (OUT):  epi: acc + w[n,kexp]*b2 (+C) -> fp32 C
// =====================================================================
template<int MODE, bool BTB>
__global__ __launch_bounds__(256)
void wmma_gemm(const __bf16* __restrict__ A, int lda, int Kd,
               const void* __restrict__ Bp, int ldB,
               const float* __restrict__ bias,
               const float* __restrict__ wgt, int kexp,
               void* __restrict__ Cptr, int ldC, int accumulate) {
  constexpr int LSTR = 40;                       // LDS pitch in halves
  __shared__ __bf16 Asl[2][128 * LSTR];          // [row][k]
  __shared__ __bf16 Bsl[2][128 * LSTR];          // [col][k] (transposed)

  const int tid  = threadIdx.x;
  const int lane = tid & 31;
  const int wave = tid >> 5;
  const int wm   = wave >> 2;                    // 0..1 -> 64-row strip
  const int wn   = wave & 3;                     // 0..3 -> 32-col strip
  const int row0 = blockIdx.y * 128;
  const int col0 = blockIdx.x * 128;
  const int half = lane >> 4;
  const int lr   = lane & 15;

  v8f acc[4][2];
#pragma unroll
  for (int mt = 0; mt < 4; ++mt)
#pragma unroll
    for (int nt = 0; nt < 2; ++nt)
#pragma unroll
      for (int v = 0; v < 8; ++v) acc[mt][nt][v] = 0.0f;

  // ---- staging helpers (pure copies on the bf16 paths) ----
  auto loadA = [&](int k0, bf4_t ar[4]) {
#pragma unroll
    for (int i = 0; i < 4; ++i) {
      const int ch = tid + i * 256;              // 128 rows x 8 k-chunks
      const int r = ch >> 3, c4 = (ch & 7) << 2;
      const __bf16* src = A + (long)(row0 + r) * lda + (k0 + c4);
      if (i == 0 && k0 + 64 < Kd) __builtin_prefetch(src + 64, 0, 1);
      ar[i] = *(const bf4_t*)src;                // global_load_b64
    }
  };
  auto storeA = [&](int buf, const bf4_t ar[4]) {
#pragma unroll
    for (int i = 0; i < 4; ++i) {
      const int ch = tid + i * 256;
      const int r = ch >> 3, c4 = (ch & 7) << 2;
      *(bf4_t*)(&Asl[buf][r * LSTR + c4]) = ar[i];   // ds_store_b64
    }
  };
  auto loadB_bt = [&](int k0, bf4_t br[4]) {
    const __bf16* BT = (const __bf16*)Bp;        // [Ndim][Kd]
#pragma unroll
    for (int i = 0; i < 4; ++i) {
      const int ch = tid + i * 256;              // 128 cols x 8 k-chunks
      const int c = ch >> 3, c4 = (ch & 7) << 2;
      br[i] = *(const bf4_t*)(BT + (long)(col0 + c) * Kd + (k0 + c4));
    }
  };
  auto storeB_bt = [&](int buf, const bf4_t br[4]) {
#pragma unroll
    for (int i = 0; i < 4; ++i) {
      const int ch = tid + i * 256;
      const int c = ch >> 3, c4 = (ch & 7) << 2;
      *(bf4_t*)(&Bsl[buf][c * LSTR + c4]) = br[i];
    }
  };
  auto loadB_f32 = [&](int k0, f4_t bw[4]) {
#pragma unroll
    for (int i = 0; i < 4; ++i) {
      const int ch = tid + i * 256;              // 32 krows x 32 col-chunks
      const int kr = ch >> 5, c4 = (ch & 31) << 2;
      bw[i] = *(const f4_t*)((const float*)Bp + (long)(k0 + kr) * ldB + col0 + c4);
    }
  };
  auto storeB_f32 = [&](int buf, const f4_t bw[4]) {
#pragma unroll
    for (int i = 0; i < 4; ++i) {
      const int ch = tid + i * 256;
      const int kr = ch >> 5, c4 = (ch & 31) << 2;
#pragma unroll
      for (int j = 0; j < 4; ++j)
        Bsl[buf][(c4 + j) * LSTR + kr] = f2bf(bw[i][j]);
    }
  };

  // ---- prologue: stage tile 0 ----
  {
    bf4_t ar[4];
    loadA(0, ar);
    storeA(0, ar);
    if constexpr (BTB) {
      bf4_t br[4]; loadB_bt(0, br); storeB_bt(0, br);
    } else {
      f4_t bw[4]; loadB_f32(0, bw); storeB_f32(0, bw);
    }
  }
  __syncthreads();

  int cur = 0;
  for (int k0 = 0; k0 < Kd; k0 += 32) {
    const bool more = (k0 + 32) < Kd;
    // issue next tile's global loads before compute (latency overlap)
    bf4_t arn[4], brn[4];
    f4_t bwn[4];
    if (more) {
      loadA(k0 + 32, arn);
      if constexpr (BTB) loadB_bt(k0 + 32, brn);
      else               loadB_f32(k0 + 32, bwn);
    }

    // ---- fragments per ISA 16-bit wave32 layouts ----
    v16bf af[4], bfr[2];
#pragma unroll
    for (int mt = 0; mt < 4; ++mt) {
      // A 16x32: lanes0-15 K {0..7,16..23}, lanes16-31 K {8..15,24..31}
      const __bf16* p = &Asl[cur][(wm * 64 + mt * 16 + lr) * LSTR + half * 8];
      bf8_t lo = *(const bf8_t*)p;               // ds_load_b128
      bf8_t hi = *(const bf8_t*)(p + 16);
#pragma unroll
      for (int e = 0; e < 8; ++e) { af[mt][e] = lo[e]; af[mt][e + 8] = hi[e]; }
    }
#pragma unroll
    for (int nt = 0; nt < 2; ++nt) {
      // B 32x16: lanes0-15 K {0..7,16..23}, lanes16-31 K {8..15,24..31} (col per lane)
      const __bf16* p = &Bsl[cur][(wn * 32 + nt * 16 + lr) * LSTR + half * 8];
      bf8_t lo = *(const bf8_t*)p;
      bf8_t hi = *(const bf8_t*)(p + 16);
#pragma unroll
      for (int e = 0; e < 8; ++e) { bfr[nt][e] = lo[e]; bfr[nt][e + 8] = hi[e]; }
    }

#pragma unroll
    for (int mt = 0; mt < 4; ++mt)
#pragma unroll
      for (int nt = 0; nt < 2; ++nt)
        acc[mt][nt] = __builtin_amdgcn_wmma_f32_16x16x32_bf16(
            false, af[mt], false, bfr[nt], (short)0, acc[mt][nt], false, false);

    if (more) {
      storeA(cur ^ 1, arn);
      if constexpr (BTB) storeB_bt(cur ^ 1, brn);
      else               storeB_f32(cur ^ 1, bwn);
    }
    __syncthreads();                             // one barrier per k-step
    cur ^= 1;
  }

  // ---- epilogue: C/D 16x16 f32 layout: row = half*8 + v, col = lr ----
#pragma unroll
  for (int mt = 0; mt < 4; ++mt) {
    const int rbase = row0 + wm * 64 + mt * 16 + half * 8;
#pragma unroll
    for (int nt = 0; nt < 2; ++nt) {
      const int col = col0 + wn * 32 + nt * 16 + lr;
#pragma unroll
      for (int v = 0; v < 8; ++v) {
        const int grow = rbase + v;
        float r = acc[mt][nt][v];
        if (MODE == 0) {
          r += bias[col];
          ((float*)Cptr)[(long)grow * ldC + col] = r;
        } else if (MODE == 1) {
          r = gelu_exact(r + bias[col]);
          r *= wgt[grow * KEXP + kexp];          // fold softmax weight into hid
          ((__bf16*)Cptr)[(long)grow * ldC + col] = f2bf(r);
        } else {
          r += wgt[grow * KEXP + kexp] * bias[col];
          float* O = (float*)Cptr;
          const long idx = (long)grow * ldC + col;
          if (accumulate) r += O[idx];
          O[idx] = r;
        }
      }
    }
  }
}

// ========== pack cond = concat(h_anchor, h_mask) -> bf16 [N,8192] ==========
__global__ __launch_bounds__(256)
void pack_cond(const float* __restrict__ anchor, const float* __restrict__ mask,
               __bf16* __restrict__ cond) {
  const long base = (long)(blockIdx.x * 256 + threadIdx.x) * 8;
  const long n = base >> 13;                     // / 8192
  const int  d = (int)(base & 8191);
  const float* src = (d < D_MODEL) ? anchor + n * D_MODEL + d
                                   : mask   + n * D_MODEL + (d - D_MODEL);
  f4_t v0 = *(const f4_t*)src;
  f4_t v1 = *(const f4_t*)(src + 4);
  bf8_t o;
  o[0] = f2bf(v0[0]); o[1] = f2bf(v0[1]); o[2] = f2bf(v0[2]); o[3] = f2bf(v0[3]);
  o[4] = f2bf(v1[0]); o[5] = f2bf(v1[1]); o[6] = f2bf(v1[2]); o[7] = f2bf(v1[3]);
  *(bf8_t*)(cond + base) = o;                    // global_store_b128
}

// ====== transpose+convert: src fp32 [R][C] (per z) -> dst bf16 [C][R] ======
__global__ __launch_bounds__(256)
void transpose_bf16(const float* __restrict__ src, __bf16* __restrict__ dst,
                    int R, int C) {
  __shared__ float t[32][33];
  const long zoff = (long)blockIdx.z * R * C;
  const float* s = src + zoff;
  __bf16* d = dst + zoff;
  const int r0 = blockIdx.y * 32, c0 = blockIdx.x * 32;
  const int tx = threadIdx.x, ty = threadIdx.y;  // block (32,8)
#pragma unroll
  for (int i = 0; i < 4; ++i)
    t[ty + i * 8][tx] = s[(long)(r0 + ty + i * 8) * C + c0 + tx];
  __syncthreads();
#pragma unroll
  for (int i = 0; i < 4; ++i)
    d[(long)(c0 + ty + i * 8) * R + r0 + tx] = f2bf(t[tx][ty + i * 8]);
}

// ===================== routing softmax: weights[N,8] =====================
__global__ __launch_bounds__(256)
void routing_kernel(const float* __restrict__ hm, const float* __restrict__ Wr,
                    const float* __restrict__ br, float* __restrict__ wout) {
  const int row  = blockIdx.x * 8 + (threadIdx.x >> 5);
  const int lane = threadIdx.x & 31;
  const float* h = hm + (long)row * D_MODEL;
  float a[KEXP];
#pragma unroll
  for (int j = 0; j < KEXP; ++j) a[j] = 0.0f;
  for (int d = lane; d < D_MODEL; d += 32) {
    const float x = h[d];
    const float* wr = Wr + d * KEXP;
#pragma unroll
    for (int j = 0; j < KEXP; ++j) a[j] += x * wr[j];
  }
#pragma unroll
  for (int j = 0; j < KEXP; ++j)
#pragma unroll
    for (int off = 16; off > 0; off >>= 1) a[j] += __shfl_down(a[j], off, 32);
  if (lane == 0) {
#pragma unroll
    for (int j = 0; j < KEXP; ++j) a[j] += br[j];
    float m = a[0];
#pragma unroll
    for (int j = 1; j < KEXP; ++j) m = fmaxf(m, a[j]);
    float s = 0.0f, e[KEXP];
#pragma unroll
    for (int j = 0; j < KEXP; ++j) { e[j] = expf(a[j] - m); s += e[j]; }
    const float inv = 1.0f / s;
#pragma unroll
    for (int j = 0; j < KEXP; ++j) wout[row * KEXP + j] = e[j] * inv;
  }
}

// ========= relevance gate sigmoid(q.k/sqrt(512)) + in-place scale =========
__global__ __launch_bounds__(256)
void relscale_kernel(const float* __restrict__ q, const float* __restrict__ k,
                     float* __restrict__ out) {
  const int n = blockIdx.x;
  __shared__ float red[8];
  __shared__ float relsh;
  const int tid = threadIdx.x, lane = tid & 31, wv = tid >> 5;
  const float* qp = q + (long)n * D_PROJ;
  const float* kp = k + (long)n * D_PROJ;
  float p = 0.0f;
  for (int j = tid; j < D_PROJ; j += 256) p += qp[j] * kp[j];
#pragma unroll
  for (int off = 16; off > 0; off >>= 1) p += __shfl_down(p, off, 32);
  if (lane == 0) red[wv] = p;
  __syncthreads();
  if (tid == 0) {
    float s = 0.0f;
#pragma unroll
    for (int i = 0; i < 8; ++i) s += red[i];
    const float x = s * 0.04419417382415922f;    // 1/sqrt(512)
    relsh = 1.0f / (1.0f + expf(-x));
  }
  __syncthreads();
  const float rel = relsh;
  f4_t* o = (f4_t*)(out + (long)n * D_MODEL);
#pragma unroll
  for (int i = 0; i < 4; ++i) {
    const int j = tid + i * 256;
    f4_t v = o[j];
    v[0] *= rel; v[1] *= rel; v[2] *= rel; v[3] *= rel;
    o[j] = v;
  }
}

extern "C" void kernel_launch(void* const* d_in, const int* in_sizes, int n_in,
                              void* d_out, int out_size, void* d_ws, size_t ws_size,
                              hipStream_t stream) {
  const float* h_anchor = (const float*)d_in[0];
  const float* h_mask   = (const float*)d_in[1];
  const float* Wr       = (const float*)d_in[2];
  const float* br       = (const float*)d_in[3];
  const float* W1       = (const float*)d_in[4];
  const float* b1       = (const float*)d_in[5];
  const float* W2       = (const float*)d_in[6];
  const float* b2       = (const float*)d_in[7];
  const float* Wq       = (const float*)d_in[8];
  const float* bq       = (const float*)d_in[9];
  const float* Wk       = (const float*)d_in[10];
  const float* bk       = (const float*)d_in[11];
  float* out = (float*)d_out;

  // ---- workspace layout ----
  char* ws = (char*)d_ws;
  size_t off = 0;
  float* w_wgt = (float*)(ws + off);  off += (size_t)NTOK * KEXP * 4;        // 128KB
  float* w_q   = (float*)(ws + off);  off += (size_t)NTOK * D_PROJ * 4;      // 8MB
  float* w_k   = (float*)(ws + off);  off += (size_t)NTOK * D_PROJ * 4;      // 8MB
  __bf16* w_cond = (__bf16*)(ws + off); off += (size_t)NTOK * 2 * D_MODEL * 2; // 64MB
  __bf16* w_hid  = (__bf16*)(ws + off); off += (size_t)NTOK * D_HID * 2;     // 16MB
  // optional pre-transposed bf16 weights (~410MB)
  __bf16* w_WqT = (__bf16*)(ws + off); off += (size_t)D_MODEL * D_PROJ * 2;
  __bf16* w_WkT = (__bf16*)(ws + off); off += (size_t)D_MODEL * D_PROJ * 2;
  __bf16* w_W1T = (__bf16*)(ws + off); off += (size_t)KEXP * 2 * D_MODEL * D_HID * 2;
  __bf16* w_W2T = (__bf16*)(ws + off); off += (size_t)KEXP * D_HID * D_MODEL * 2;
  const bool big = (ws_size >= off);  // enough room for bf16 weight images?

  // 0) one-shot bf16 packing / weight transposition
  pack_cond<<<(NTOK * 2 * D_MODEL) / (256 * 8), 256, 0, stream>>>(h_anchor, h_mask, w_cond);
  if (big) {
    dim3 blk(32, 8);
    transpose_bf16<<<dim3(D_PROJ / 32, D_MODEL / 32, 1), blk, 0, stream>>>(Wq, w_WqT, D_MODEL, D_PROJ);
    transpose_bf16<<<dim3(D_PROJ / 32, D_MODEL / 32, 1), blk, 0, stream>>>(Wk, w_WkT, D_MODEL, D_PROJ);
    transpose_bf16<<<dim3(D_HID / 32, (2 * D_MODEL) / 32, KEXP), blk, 0, stream>>>(W1, w_W1T, 2 * D_MODEL, D_HID);
    transpose_bf16<<<dim3(D_MODEL / 32, D_HID / 32, KEXP), blk, 0, stream>>>(W2, w_W2T, D_HID, D_MODEL);
  }

  // 1) routing softmax weights
  routing_kernel<<<NTOK / 8, 256, 0, stream>>>(h_mask, Wr, br, w_wgt);

  // 2) gate projections: q = h_mask@Wq+bq, k = h_anchor@Wk+bk
  const __bf16* A_anchor = w_cond;                 // cols [0,4096)
  const __bf16* A_mask   = w_cond + D_MODEL;       // cols [4096,8192)
  const dim3 gProj(D_PROJ / 128, NTOK / 128);
  if (big) {
    wmma_gemm<0, true><<<gProj, 256, 0, stream>>>(A_mask, 2 * D_MODEL, D_MODEL,
        w_WqT, 0, bq, nullptr, 0, w_q, D_PROJ, 0);
    wmma_gemm<0, true><<<gProj, 256, 0, stream>>>(A_anchor, 2 * D_MODEL, D_MODEL,
        w_WkT, 0, bk, nullptr, 0, w_k, D_PROJ, 0);
  } else {
    wmma_gemm<0, false><<<gProj, 256, 0, stream>>>(A_mask, 2 * D_MODEL, D_MODEL,
        Wq, D_PROJ, bq, nullptr, 0, w_q, D_PROJ, 0);
    wmma_gemm<0, false><<<gProj, 256, 0, stream>>>(A_anchor, 2 * D_MODEL, D_MODEL,
        Wk, D_PROJ, bk, nullptr, 0, w_k, D_PROJ, 0);
  }

  // 3) per-expert bottleneck MLP, softmax-weighted, accumulated into d_out
  const dim3 gHid(D_HID / 128, NTOK / 128);
  const dim3 gOut(D_MODEL / 128, NTOK / 128);
  for (int k = 0; k < KEXP; ++k) {
    const float* b1k = b1 + (long)k * D_HID;
    const float* b2k = b2 + (long)k * D_MODEL;
    if (big) {
      const __bf16* W1Tk = w_W1T + (long)k * 2 * D_MODEL * D_HID;  // [2048][8192]
      const __bf16* W2Tk = w_W2T + (long)k * D_HID * D_MODEL;      // [4096][2048]
      wmma_gemm<1, true><<<gHid, 256, 0, stream>>>(w_cond, 2 * D_MODEL, 2 * D_MODEL,
          W1Tk, 0, b1k, w_wgt, k, w_hid, D_HID, 0);
      wmma_gemm<2, true><<<gOut, 256, 0, stream>>>(w_hid, D_HID, D_HID,
          W2Tk, 0, b2k, w_wgt, k, out, D_MODEL, (k > 0) ? 1 : 0);
    } else {
      const float* W1k = W1 + (long)k * 2 * D_MODEL * D_HID;
      const float* W2k = W2 + (long)k * D_HID * D_MODEL;
      wmma_gemm<1, false><<<gHid, 256, 0, stream>>>(w_cond, 2 * D_MODEL, 2 * D_MODEL,
          W1k, D_HID, b1k, w_wgt, k, w_hid, D_HID, 0);
      wmma_gemm<2, false><<<gOut, 256, 0, stream>>>(w_hid, D_HID, D_HID,
          W2k, D_MODEL, b2k, w_wgt, k, out, D_MODEL, (k > 0) ? 1 : 0);
    }
  }

  // 4) relevance sigmoid gate, in-place scale of out
  relscale_kernel<<<NTOK, 256, 0, stream>>>(w_q, w_k, out);
}
